// GrphConvNetwork_23587960389876
// MI455X (gfx1250) — matrix operand order repaired and checked by reading, not compiled, not run on hardware
//
#include <hip/hip_runtime.h>

// ---------------------------------------------------------------------------
// GCN (GraphConv, norm='both') x 8 layers on MI455X / gfx1250.
//   per layer: h*=deg_out^-1/2 ; agg = scatter_add(h[src] -> dst) ;
//              agg*=deg_in^-1/2 ; out = agg@W + b ; LayerNorm ; erf-GELU
// GEMM runs on v_wmma_f32_16x16x32_bf16 (bf16 operands, f32 accumulate).
// Feature matrix (102.4 MB) fits MI455X's 192 MB L2, so the edge gather /
// scatter-atomics are L2 traffic; HBM sees ~0.3 GB/layer -> memory bound.
// M-tile = 32 rows/WG so each B fragment (Wt, L2-resident) is reused by two
// A row-tiles: 8 WMMAs per K-step per wave, halving L2 B-traffic vs M=16.
// ---------------------------------------------------------------------------

#define N_NODES  50000
#define N_EDGES  160000
#define DIM      512
#define N_LAYERS 8
#define LN_EPS   1e-5f

typedef __bf16 bf16_t;
typedef __attribute__((ext_vector_type(16))) __bf16 v16bf;
typedef __attribute__((ext_vector_type(8)))  __bf16 v8bf;
typedef __attribute__((ext_vector_type(8)))  float  v8f;

union ABFrag { v16bf v; v8bf h[2]; };

// ---------------------------------------------------------------- norms ----
__global__ void init_norms(float* __restrict__ nout, float* __restrict__ nin) {
    int i = blockIdx.x * 256 + threadIdx.x;
    if (i < N_NODES) { nout[i] = 0.f; nin[i] = 0.f; }
}

__global__ void degree_kernel(const int* __restrict__ src, const int* __restrict__ dst,
                              float* __restrict__ nout, float* __restrict__ nin) {
    int e = blockIdx.x * 256 + threadIdx.x;   // grid sized exactly E/256
    atomicAdd(&nout[src[e]], 1.0f);
    atomicAdd(&nin[dst[e]], 1.0f);
}

__global__ void finalize_norms(float* __restrict__ nout, float* __restrict__ nin) {
    int i = blockIdx.x * 256 + threadIdx.x;
    if (i < N_NODES) {
        nout[i] = rsqrtf(fmaxf(nout[i], 1.0f));
        nin[i]  = rsqrtf(fmaxf(nin[i], 1.0f));
    }
}

// --------------------------------------------- weights: fp32 W -> bf16 W^T --
// Wt[l][n][k] = bf16(W[l][k][n]) : contiguous along K == WMMA B-fragment order
__global__ void convert_weights(const float* __restrict__ Ws, bf16_t* __restrict__ Wt) {
    int idx = blockIdx.x * 256 + threadIdx.x;           // < 8*512*512
    int l   = idx >> 18;
    int rem = idx & 262143;
    int n   = rem >> 9;
    int k   = rem & 511;
    Wt[idx] = (bf16_t)Ws[(l << 18) + (k << 9) + n];
}

// ----------------------------------------------------------------- SpMM ----
__global__ void zero_agg(float4* __restrict__ agg) {
    size_t i = (size_t)blockIdx.x * 256 + threadIdx.x;  // grid exactly N*D/4/256
    agg[i] = make_float4(0.f, 0.f, 0.f, 0.f);
}

// one thread = one (edge, 16-byte chunk); wave covers 512 B of one edge row
__global__ void edge_scatter(const float* __restrict__ h,
                             const int* __restrict__ src, const int* __restrict__ dst,
                             const float* __restrict__ norm_out,
                             float* __restrict__ agg) {
    size_t idx = (size_t)blockIdx.x * 256 + threadIdx.x; // < E * 128
    int e = (int)(idx >> 7);
    int c = (int)(idx & 127);
    int s = src[e];
    int d = dst[e];
    float4 v = ((const float4*)h)[(size_t)s * 128 + c];
    float sc = norm_out[s];
    float* o = agg + (size_t)d * DIM + c * 4;
    atomicAdd(o + 0, v.x * sc);
    atomicAdd(o + 1, v.y * sc);
    atomicAdd(o + 2, v.z * sc);
    atomicAdd(o + 3, v.w * sc);
}

// ------------------------------------- fused GEMM + bias + LN + GELU -------
// WG = 256 threads (8 waves). Block computes rows [m0, m0+32) x all 512 cols.
// Wave w owns cols [w*64, w*64+64): 2 M-tiles x 4 N-tiles = 8 accumulators;
// B fragments reused across both M-tiles (halves L2 traffic for Wt).
__global__ __launch_bounds__(256) void gemm_ln_gelu(
    const float* __restrict__ agg, const float* __restrict__ norm_in,
    const bf16_t* __restrict__ Wt, const float* __restrict__ bias,
    const float* __restrict__ g, const float* __restrict__ beta,
    float* __restrict__ out)
{
    __shared__ __align__(16) bf16_t sA[32 * DIM];   // 32 KB: A strip, bf16
    __shared__ __align__(16) float  sC[32 * DIM];   // 64 KB: C strip, f32
    __shared__ float sSum[256], sSq[256];
    __shared__ float sMu[32], sInv[32];

    const int tid = threadIdx.x;
    const int m0  = blockIdx.x * 32;                // 1563 blocks, last partial

    // ---- stage A = bf16(agg[row] * norm_in[row]) into LDS -----------------
    #pragma unroll
    for (int j = 0; j < 16; ++j) {
        int i4  = tid + j * 256;                    // 0..4095 float4 slots
        int row = i4 >> 7;
        int c4  = i4 & 127;
        int gr  = m0 + row;
        float4 v = make_float4(0.f, 0.f, 0.f, 0.f);
        float  s = 0.f;
        if (gr < N_NODES) {
            v = ((const float4*)agg)[(size_t)gr * 128 + c4];
            s = norm_in[gr];
        }
        bf16_t* p = &sA[row * DIM + c4 * 4];
        p[0] = (bf16_t)(v.x * s);
        p[1] = (bf16_t)(v.y * s);
        p[2] = (bf16_t)(v.z * s);
        p[3] = (bf16_t)(v.w * s);
    }
    __syncthreads();

    const int lane = tid & 31;
    const int wid  = tid >> 5;
    const int n0w  = wid * 64;
    const int lrow = lane & 15;       // A row / B column within 16x16 tile
    const int lhi  = lane >> 4;       // half-wave selector for K striping

    v8f acc[2][4] = {};

    #pragma unroll
    for (int k0 = 0; k0 < DIM; k0 += 32) {
        // A 16x32 bf16 fragments (ISA 7.12.2): lane<16 -> K k0+0..7 & k0+16..23
        //                                      lane>=16 -> K k0+8..15 & k0+24..31
        ABFrag a[2];
        #pragma unroll
        for (int m = 0; m < 2; ++m) {
            const bf16_t* ap = &sA[(m * 16 + lrow) * DIM + k0 + lhi * 8];
            a[m].h[0] = *(const v8bf*)(ap);
            a[m].h[1] = *(const v8bf*)(ap + 16);
        }
        #pragma unroll
        for (int t = 0; t < 4; ++t) {
            // B 32x16 fragment from W^T: lane&15 -> column, K striped like A
            const bf16_t* bp = Wt + (size_t)(n0w + t * 16 + lrow) * DIM + k0 + lhi * 8;
            ABFrag b;
            b.h[0] = *(const v8bf*)(bp);
            b.h[1] = *(const v8bf*)(bp + 16);
            acc[0][t] = __builtin_amdgcn_wmma_f32_16x16x32_bf16(
                false, a[0].v, false, b.v, (short)0, acc[0][t], false, false);
            acc[1][t] = __builtin_amdgcn_wmma_f32_16x16x32_bf16(
                false, a[1].v, false, b.v, (short)0, acc[1][t], false, false);
        }
    }

    // ---- spill C + bias to LDS (C layout: VGPR r -> M=r+8*(lane>=16), N=lane&15)
    #pragma unroll
    for (int t = 0; t < 4; ++t) {
        int   n  = n0w + t * 16 + lrow;
        float bv = bias[n];
        #pragma unroll
        for (int m = 0; m < 2; ++m) {
            #pragma unroll
            for (int r = 0; r < 8; ++r) {
                int row = m * 16 + r + lhi * 8;
                sC[row * DIM + n] = acc[m][t][r] + bv;
            }
        }
    }
    __syncthreads();

    // ---- LayerNorm + erf-GELU: 8 threads per row, 64 cols each ------------
    const int row = tid >> 3;
    const int sub = tid & 7;
    const float* cr = &sC[row * DIM + sub * 64];
    float sum = 0.f, sq = 0.f;
    #pragma unroll
    for (int i = 0; i < 64; ++i) { float x = cr[i]; sum += x; sq += x * x; }
    sSum[tid] = sum; sSq[tid] = sq;
    __syncthreads();
    if (sub == 0) {
        float ts = 0.f, tq = 0.f;
        #pragma unroll
        for (int i = 0; i < 8; ++i) { ts += sSum[row * 8 + i]; tq += sSq[row * 8 + i]; }
        float mu  = ts * (1.0f / DIM);
        float var = tq * (1.0f / DIM) - mu * mu;
        sMu[row]  = mu;
        sInv[row] = rsqrtf(var + LN_EPS);
    }
    __syncthreads();
    if (m0 + row < N_NODES) {
        float mu  = sMu[row];
        float inv = sInv[row];
        float* orow = out + (size_t)(m0 + row) * DIM + sub * 64;
        #pragma unroll
        for (int i = 0; i < 64; ++i) {
            int   col = sub * 64 + i;
            float y   = (cr[i] - mu) * inv * g[col] + beta[col];
            orow[i]   = 0.5f * y * (1.0f + erff(y * 0.70710678118654752f));
        }
    }
}

// ---------------------------------------------------------------------------
extern "C" void kernel_launch(void* const* d_in, const int* in_sizes, int n_in,
                              void* d_out, int out_size, void* d_ws, size_t ws_size,
                              hipStream_t stream) {
    const float* features = (const float*)d_in[0];
    const int*   src      = (const int*)d_in[1];
    const int*   dst      = (const int*)d_in[2];
    const float* Ws       = (const float*)d_in[3];
    const float* bs       = (const float*)d_in[4];
    const float* ln_g     = (const float*)d_in[5];
    const float* ln_b     = (const float*)d_in[6];
    float*       outp     = (float*)d_out;

    const size_t featBytes = (size_t)N_NODES * DIM * sizeof(float);   // 102.4 MB
    char* w = (char*)d_ws;
    float*  bufA = (float*)w;  w += featBytes;                        // ping buffer
    float*  agg  = (float*)w;  w += featBytes;                        // SpMM output
    bf16_t* Wt   = (bf16_t*)w; w += (size_t)N_LAYERS * DIM * DIM * sizeof(bf16_t);
    float*  nout = (float*)w;  w += 200192;                           // rsqrt(deg_out)
    float*  nin  = (float*)w;  w += 200192;                           // rsqrt(deg_in)

    // one-time per launch (deterministic; no caching across calls)
    init_norms    <<<(N_NODES + 255) / 256, 256, 0, stream>>>(nout, nin);
    degree_kernel <<<N_EDGES / 256,        256, 0, stream>>>(src, dst, nout, nin);
    finalize_norms<<<(N_NODES + 255) / 256, 256, 0, stream>>>(nout, nin);
    convert_weights<<<(N_LAYERS * DIM * DIM) / 256, 256, 0, stream>>>(Ws, Wt);

    const float* hin = features;
    for (int l = 0; l < N_LAYERS; ++l) {
        float* hout = (l & 1) ? outp : bufA;      // layer 7 (odd) -> d_out
        zero_agg    <<<(N_NODES * DIM / 4) / 256,     256, 0, stream>>>((float4*)agg);
        edge_scatter<<<((size_t)N_EDGES * 128) / 256, 256, 0, stream>>>(
            hin, src, dst, nout, agg);
        gemm_ln_gelu<<<(N_NODES + 31) / 32, 256, 0, stream>>>(
            agg, nin, Wt + (size_t)l * DIM * DIM,
            bs + l * DIM, ln_g + l * DIM, ln_b + l * DIM, hout);
        hin = hout;
    }
}
